// Word2Vec_38079180046938
// MI455X (gfx1250) — matrix operand order repaired and checked by reading, not compiled, not run on hardware
//
#include <hip/hip_runtime.h>
#include <hip/hip_bf16.h>

typedef __attribute__((ext_vector_type(16))) __bf16 v16bf;
typedef __attribute__((ext_vector_type(8)))  __bf16 v8bf;
typedef __attribute__((ext_vector_type(8)))  float  v8f;

#define L_TOK   4096
#define VOCAB   32000
#define DIM     128
#define WIN     5
#define NCTR    (L_TOK - 2*WIN)          // 4086 centers
#define NPAD    4096                     // padded rows (multiple of 16)
#define NCHUNK  10                       // column chunks
#define CPC     (VOCAB / NCHUNK)         // 3200 cols per chunk
#define NTILES  (CPC / 16)               // 200 N-tiles per chunk (even)
#define LOG2E   1.4426950408889634f
#define LN2     0.6931471805599453f

// Raw hardware transcendentals: v_exp_f32 (2^x) and v_log_f32 (log2 x).
// Arguments to exp2 are always <= 0 here; underflow-to-zero is exactly what
// an online softmax wants, so the libm denormal guard is pure overhead.
__device__ __forceinline__ float exp2_raw(float x) { return __builtin_amdgcn_exp2f(x); }
__device__ __forceinline__ float log2_raw(float x) { return __builtin_amdgcn_logf(x); }

// Build one 16-half WMMA fragment from two contiguous 16-byte chunks.
__device__ __forceinline__ v16bf load_frag(const __bf16* p, int gap) {
  v8bf lo = *reinterpret_cast<const v8bf*>(p);
  v8bf hi = *reinterpret_cast<const v8bf*>(p + gap);
  return __builtin_shufflevector(lo, hi, 0,1,2,3,4,5,6,7,8,9,10,11,12,13,14,15);
}

// ---------------------------------------------------------------------------
// Kernel 1: ctx_mean (fp32 accumulate) -> bf16 A matrix, zero-padded rows.
// ---------------------------------------------------------------------------
__global__ void k_ctx_mean(const int* __restrict__ tokens,
                           const float* __restrict__ emb,
                           __bf16* __restrict__ A) {
  const int i = blockIdx.x;        // padded row 0..4095
  const int d = threadIdx.x;       // 0..127
  float s = 0.0f;
  if (i < NCTR) {
    const int center = i + WIN;
#pragma unroll
    for (int j = -WIN; j <= WIN; ++j) {
      if (j == 0) continue;
      const int tok = tokens[center + j];
      s += emb[(long)tok * DIM + d];
    }
    s *= (1.0f / (2 * WIN));
  }
  A[(long)i * DIM + d] = (__bf16)s;
}

// ---------------------------------------------------------------------------
// Kernel 2: W fp32 -> bf16 row-major copy.
// ---------------------------------------------------------------------------
__global__ void k_w_bf16(const float* __restrict__ Wf, __bf16* __restrict__ Wb) {
  const long idx = (long)blockIdx.x * blockDim.x + threadIdx.x;
  if (idx < (long)VOCAB * DIM) Wb[idx] = (__bf16)Wf[idx];
}

// ---------------------------------------------------------------------------
// Kernel 3: fused bf16 WMMA GEMM + online softmax partials.
// grid = (32, NCHUNK), block = 256 (8 waves). wave -> one 16-row M tile.
// Two-stage software pipeline at single-tile granularity (4 fragments =
// 32 VGPRs per stage): the next tile's load clause issues before the current
// tile's 4 WMMAs + softmax update, hiding load latency without spilling.
// stats[(row*NCHUNK + chunk)] = {running max (base-2), running sumexp2}
// ---------------------------------------------------------------------------
__global__ void k_gemm_softmax(const __bf16* __restrict__ A,
                               const __bf16* __restrict__ Wb,
                               float2* __restrict__ stats) {
  const int lane  = threadIdx.x & 31;
  const int wave  = threadIdx.x >> 5;
  const int mtile = blockIdx.x * 8 + wave;        // 0..255
  const int chunk = blockIdx.y;                   // 0..NCHUNK-1

  // ---- A fragments: ISA 16-bit A 16x32 layout.
  // lanes 0-15: halves[0..7]=K(s*32+0..7), halves[8..15]=K(s*32+16..23)
  // lanes 16-31: +8 on both K bases.
  const int row = mtile * 16 + (lane & 15);
  const int hiA = (lane >= 16) ? 8 : 0;
  const __bf16* arow = A + (long)row * DIM;
  v16bf afrag[4];
#pragma unroll
  for (int s = 0; s < 4; ++s)
    afrag[s] = load_frag(arow + s * 32 + hiA, 16);

  float rmax[8], rsum[8];
#pragma unroll
  for (int r = 0; r < 8; ++r) { rmax[r] = -__builtin_inff(); rsum[r] = 0.0f; }

  // ---- B fragments: ISA 16-bit B 32x16 layout (lane = column, contiguous K
  // run per half-wave: halves[0..7]=K hiB+0..7, halves[8..15]=K hiB+16..23, lanes 16-31 -> hiB=8).
  const int hiB = (lane >= 16) ? 8 : 0;
  const int cln = lane & 15;
  const __bf16* bbase = Wb + (long)(chunk * CPC + cln) * DIM + hiB;

  // Load the 4 fragments of one 16-col tile at tile index tp.
  auto load4 = [&](int tp, v16bf* dst) {
#pragma unroll
    for (int s = 0; s < 4; ++s)
      dst[s] = load_frag(bbase + (long)tp * 16 * DIM + s * 32, 16);
  };

  // 4 WMMAs (one accumulation chain) + online softmax update for one tile.
  auto compute4 = [&](const v16bf* bfr) {
    v8f c = {};
#pragma unroll
    for (int s = 0; s < 4; ++s)
      c = __builtin_amdgcn_wmma_f32_16x16x32_bf16(
              false, afrag[s], false, bfr[s], (short)0, c, false, false);
    // C layout: VGPR r / lane l: lanes 0-15 -> (M=r, N=l),
    // lanes 16-31 -> (M=8+r, N=l-16).
#pragma unroll
    for (int r = 0; r < 8; ++r) {
      const float v  = c[r] * LOG2E;
      const float nm = fmaxf(rmax[r], v);
      rsum[r] = rsum[r] * exp2_raw(rmax[r] - nm) + exp2_raw(v - nm);
      rmax[r] = nm;
    }
  };

  v16bf buf0[4], buf1[4];
  load4(0, buf0);
  for (int t = 0; t < NTILES; t += 2) {
    load4(t + 1, buf1);                            // next tile into other buffer
    __builtin_prefetch(bbase + (long)(t + 2) * 16 * DIM, 0, 1);
    compute4(buf0);
    if (t + 2 < NTILES) load4(t + 2, buf0);
    compute4(buf1);
  }

  // butterfly merge across the 16 lanes holding each row
#pragma unroll
  for (int r = 0; r < 8; ++r) {
    float m = rmax[r], s = rsum[r];
#pragma unroll
    for (int off = 1; off < 16; off <<= 1) {
      const float mo = __shfl_xor(m, off, 32);
      const float so = __shfl_xor(s, off, 32);
      const float mm = fmaxf(m, mo);
      s = s * exp2_raw(m - mm) + so * exp2_raw(mo - mm);
      m = mm;
    }
    rmax[r] = m; rsum[r] = s;
  }

  if (cln == 0) {                       // lane 0 -> rows 0..7, lane 16 -> rows 8..15
    const int rbase = mtile * 16 + ((lane >= 16) ? 8 : 0);
#pragma unroll
    for (int r = 0; r < 8; ++r)
      stats[(long)(rbase + r) * NCHUNK + chunk] = make_float2(rmax[r], rsum[r]);
  }
}

// ---------------------------------------------------------------------------
// Kernel 4: per-row loss = lse - logit[target]. One block (128 thr) per row.
// ---------------------------------------------------------------------------
__global__ void k_row_loss(const int* __restrict__ tokens,
                           const __bf16* __restrict__ A,
                           const __bf16* __restrict__ Wb,
                           const float2* __restrict__ stats,
                           float* __restrict__ rowloss) {
  const int i = blockIdx.x;                      // 0..NCTR-1
  const int tgt = tokens[i + WIN];
  __shared__ float red[DIM];
  const float p = (float)A[(long)i * DIM + threadIdx.x] *
                  (float)Wb[(long)tgt * DIM + threadIdx.x];
  red[threadIdx.x] = p;
  __syncthreads();
  for (int off = DIM / 2; off > 0; off >>= 1) {
    if (threadIdx.x < off) red[threadIdx.x] += red[threadIdx.x + off];
    __syncthreads();
  }
  if (threadIdx.x == 0) {
    float gmax = -__builtin_inff();
#pragma unroll
    for (int c = 0; c < NCHUNK; ++c) gmax = fmaxf(gmax, stats[(long)i * NCHUNK + c].x);
    float S = 0.0f;
#pragma unroll
    for (int c = 0; c < NCHUNK; ++c) {
      const float2 st = stats[(long)i * NCHUNK + c];
      S += st.y * exp2_raw(st.x - gmax);
    }
    const float lse = (gmax + log2_raw(S)) * LN2;   // natural-log lse
    rowloss[i] = lse - red[0];
  }
}

// ---------------------------------------------------------------------------
// Kernel 5: deterministic single-block final reduction.
// ---------------------------------------------------------------------------
__global__ void k_reduce(const float* __restrict__ rowloss, float* __restrict__ out) {
  __shared__ float red[256];
  float s = 0.0f;
  for (int i = threadIdx.x; i < NCTR; i += 256) s += rowloss[i];
  red[threadIdx.x] = s;
  __syncthreads();
  for (int off = 128; off > 0; off >>= 1) {
    if (threadIdx.x < off) red[threadIdx.x] += red[threadIdx.x + off];
    __syncthreads();
  }
  if (threadIdx.x == 0) out[0] = red[0];
}

// ---------------------------------------------------------------------------
extern "C" void kernel_launch(void* const* d_in, const int* in_sizes, int n_in,
                              void* d_out, int out_size, void* d_ws, size_t ws_size,
                              hipStream_t stream) {
  const int*   tokens = (const int*)d_in[0];
  const float* emb    = (const float*)d_in[1];
  const float* Wf     = (const float*)d_in[2];
  float*       out    = (float*)d_out;

  char* ws = (char*)d_ws;
  const size_t offA     = 0;                                       // 4096*128 bf16
  const size_t offWb    = offA  + (size_t)NPAD  * DIM * 2;         // 32000*128 bf16
  const size_t offStats = offWb + (size_t)VOCAB * DIM * 2;         // 4096*10 float2
  const size_t offLoss  = offStats + (size_t)NPAD * NCHUNK * 8;    // 4096 float

  __bf16* A      = (__bf16*)(ws + offA);
  __bf16* Wb     = (__bf16*)(ws + offWb);
  float2* stats  = (float2*)(ws + offStats);
  float*  rloss  = (float*)(ws + offLoss);

  k_ctx_mean<<<NPAD, DIM, 0, stream>>>(tokens, emb, A);

  const long welems = (long)VOCAB * DIM;
  k_w_bf16<<<(int)((welems + 255) / 256), 256, 0, stream>>>(Wf, Wb);

  k_gemm_softmax<<<dim3(NPAD / 16 / 8, NCHUNK), 256, 0, stream>>>(A, Wb, stats);

  k_row_loss<<<NCTR, DIM, 0, stream>>>(tokens, A, Wb, stats, rloss);

  k_reduce<<<1, 256, 0, stream>>>(rloss, out);
}
